// CrossAttention_33148557590619
// MI455X (gfx1250) — matrix operand order repaired and checked by reading, not compiled, not run on hardware
//
#include <hip/hip_runtime.h>

typedef __attribute__((ext_vector_type(16))) _Float16 v16h;
typedef __attribute__((ext_vector_type(8)))  _Float16 v8h;
typedef __attribute__((ext_vector_type(8)))  float    v8f;

#define HEADS 8
#define DHEAD 64
#define DIM   1024
#define HID   512
#define SEQ   2048
#define BATCH 2

#if __has_builtin(__builtin_amdgcn_exp2f)
#define EXP2F(x) __builtin_amdgcn_exp2f(x)
#else
#define EXP2F(x) __expf((x) * 0.69314718f)
#endif

static __device__ __forceinline__ v8f wmma_f16(v16h a, v16h b, v8f c) {
  // D = A(16x32 f16) * B(32x16 f16) + C(16x16 f32)
  return __builtin_amdgcn_wmma_f32_16x16x32_f16(
      false, a, false, b, (short)0, c, false, false);
}

// ---- WMMA fragment loaders (wave32; contiguous k = 16*hi + e, same for A and B) ----
// A (16x32, MxK): lane = {row = lane&15}; element e -> k = 16*hi+e   (hi = lane>>4)
static __device__ __forceinline__ v16h loadA_f32(const float* tile, int ld, int lane) {
  const int row = lane & 15, hi = (lane >> 4) & 1;
  const float* r = tile + (size_t)row * ld + 16 * hi;
  v16h a;
#pragma unroll
  for (int e = 0; e < 8; ++e) {
    a[e]     = (_Float16)r[e];
    a[8 + e] = (_Float16)r[8 + e];
  }
  return a;
}

static __device__ __forceinline__ v16h loadA_f16(const _Float16* tile, int ld, int lane) {
  const int row = lane & 15, hi = (lane >> 4) & 1;
  const _Float16* r = tile + (size_t)row * ld + 16 * hi;
  v8h lo = *(const v8h*)(r);        // k = 16*hi .. 16*hi+7
  v8h hh = *(const v8h*)(r + 8);    // k = 16*hi+8 .. 16*hi+15
  v16h a;
#pragma unroll
  for (int e = 0; e < 8; ++e) { a[e] = lo[e]; a[8 + e] = hh[e]; }
  return a;
}

// B (32x16, KxN): lane = {col n = lane&15}; element e -> k = 16*hi + e.
// Source is k-contiguous: src[n*ldn + k].
static __device__ __forceinline__ v16h loadB_kcontig(const _Float16* tile, int ldn, int lane) {
  const int n = lane & 15, hi = (lane >> 4) & 1;
  const _Float16* c = tile + (size_t)n * ldn + 16 * hi;
  v8h x0 = *(const v8h*)(c);
  v8h x1 = *(const v8h*)(c + 8);
  v16h b;
#pragma unroll
  for (int e = 0; e < 8; ++e) { b[e] = x0[e]; b[8 + e] = x1[e]; }
  return b;
}

// ---- Kernel 0: weight transpose + fp32 -> f16 convert: W[K][N] -> Wt[N][K] ----
__global__ void transpose_f32_to_f16(const float* __restrict__ W,
                                     _Float16* __restrict__ Wt, int K, int N) {
  int idx = blockIdx.x * blockDim.x + threadIdx.x;
  if (idx < K * N) {
    int k = idx / N, n = idx % N;
    Wt[(size_t)n * K + k] = (_Float16)W[idx];
  }
}

// ---- Kernel 1: projection GEMM  Out = X(fp32, MxK) * Wt(f16, NxK k-contig) ----
// Wave tile: 32 rows x 64 cols (B fragments feed two WMMAs each).
// LAYOUT 0: Out = Q/K head-split  (b*H + h, seq, 64)
// LAYOUT 1: Out = V transposed    (b*H + h, 64, seq)
template <int LAYOUT>
__global__ void __launch_bounds__(256)
proj_gemm(const float* __restrict__ X, const _Float16* __restrict__ Wt,
          _Float16* __restrict__ Out, int K) {
  const int lane = threadIdx.x & 31;
  const int wave = threadIdx.x >> 5;
  const int row0 = blockIdx.x * 256 + wave * 32;   // token rows
  const int col0 = blockIdx.y * 64;                // hidden cols
  const int nl = lane & 15, hi = (lane >> 4) & 1;

  v8f acc[2][4];
#pragma unroll
  for (int rt = 0; rt < 2; ++rt)
#pragma unroll
    for (int t = 0; t < 4; ++t)
#pragma unroll
      for (int e = 0; e < 8; ++e) acc[rt][t][e] = 0.0f;

  for (int kb = 0; kb < K; kb += 32) {
    v16h a0 = loadA_f32(X + (size_t)row0 * K + kb, K, lane);
    v16h a1 = loadA_f32(X + (size_t)(row0 + 16) * K + kb, K, lane);
#pragma unroll
    for (int t = 0; t < 4; ++t) {
      v16h b = loadB_kcontig(Wt + (size_t)(col0 + t * 16) * K + kb, K, lane);
      acc[0][t] = wmma_f16(a0, b, acc[0][t]);
      acc[1][t] = wmma_f16(a1, b, acc[1][t]);
    }
  }

  // C/D layout: lane col n = lane&15, rows m = 8*hi + r
#pragma unroll
  for (int rt = 0; rt < 2; ++rt) {
#pragma unroll
    for (int t = 0; t < 4; ++t) {
      const int col = col0 + t * 16 + nl;
      const int h = col >> 6, d = col & 63;
#pragma unroll
      for (int r = 0; r < 8; ++r) {
        const int g = row0 + rt * 16 + 8 * hi + r;  // global token row
        const int bb = g >> 11;                     // batch
        const int sq = g & 2047;                    // seq position
        size_t addr;
        if (LAYOUT == 0)
          addr = (size_t)(bb * HEADS + h) * (SEQ * DHEAD) + (size_t)sq * DHEAD + d;
        else
          addr = (size_t)(bb * HEADS + h) * (SEQ * DHEAD) + (size_t)d * SEQ + sq;
        Out[addr] = (_Float16)acc[rt][t][r];
      }
    }
  }
}

// ---- Kernel 2: fused flash attention. 32 queries (two 16-tiles) per wave. ----
// Computes S^T = K*Q^T so each query owns a lane column -> cheap row softmax,
// then O^T = V^T * P^T (V stored transposed), P^T rebuilt via shfl_xor(16).
// K and V A-fragments are shared across both query tiles (2x compute/byte).
__global__ void __launch_bounds__(128)
attention_kernel(const _Float16* __restrict__ Qf, const _Float16* __restrict__ Kf,
                 const _Float16* __restrict__ Vt, _Float16* __restrict__ AOut) {
  const int lane = threadIdx.x & 31;
  const int wave = threadIdx.x >> 5;
  const int bh = blockIdx.y;
  const int bb = bh >> 3, h = bh & 7;
  const int q0 = blockIdx.x * 128 + wave * 32;  // 32 queries per wave
  const int nl = lane & 15, hi = (lane >> 4) & 1;
  // softmax in exp2 domain: fold 64^-0.5 * log2(e) into the score scale
  const float kscale = 0.125f * 1.44269504f;

  const _Float16* Qbh = Qf + (size_t)bh * SEQ * DHEAD;  // (seq, 64)
  const _Float16* Kbh = Kf + (size_t)bh * SEQ * DHEAD;  // (seq, 64)
  const _Float16* Vbh = Vt + (size_t)bh * SEQ * DHEAD;  // (64, seq)

  // Preload Q^T B-fragments: B[k=d][n=query]; d = 64 -> two 32-wide k-steps.
  v16h bq[2][2];  // [query tile][k-step]
#pragma unroll
  for (int qt = 0; qt < 2; ++qt)
#pragma unroll
    for (int s = 0; s < 2; ++s) {
      const _Float16* c = Qbh + (size_t)(q0 + qt * 16 + nl) * DHEAD + 32 * s + 16 * hi;
      v8h x0 = *(const v8h*)(c);
      v8h x1 = *(const v8h*)(c + 8);
#pragma unroll
      for (int e = 0; e < 8; ++e) { bq[qt][s][e] = x0[e]; bq[qt][s][8 + e] = x1[e]; }
    }

  v8f o[2][4];  // O^T accumulators per query tile, 4 d-tiles x 16 queries
#pragma unroll
  for (int qt = 0; qt < 2; ++qt)
#pragma unroll
    for (int t = 0; t < 4; ++t)
#pragma unroll
      for (int e = 0; e < 8; ++e) o[qt][t][e] = 0.0f;
  float m_run[2] = {-1e30f, -1e30f};
  float l_run[2] = {0.0f, 0.0f};

  for (int kb = 0; kb < SEQ; kb += 32) {
    // K A-fragments, shared by both query tiles: [key tile][k-step]
    v16h ka[2][2];
#pragma unroll
    for (int kt = 0; kt < 2; ++kt)
#pragma unroll
      for (int s = 0; s < 2; ++s)
        ka[kt][s] = loadA_f16(Kbh + (size_t)(kb + 16 * kt) * DHEAD + 32 * s, DHEAD, lane);

    v16h pb[2];  // P^T B-fragments per query tile
#pragma unroll
    for (int qt = 0; qt < 2; ++qt) {
      // S^T tiles: st0 = keys kb..kb+15, st1 = keys kb+16..kb+31
      v8f st0, st1;
#pragma unroll
      for (int e = 0; e < 8; ++e) { st0[e] = 0.0f; st1[e] = 0.0f; }
#pragma unroll
      for (int s = 0; s < 2; ++s) {
        st0 = wmma_f16(ka[0][s], bq[qt][s], st0);
        st1 = wmma_f16(ka[1][s], bq[qt][s], st1);
      }

      // Online softmax for this lane's query column (exp2 domain).
      float p0[8], p1[8];
      float bmax = -1e30f;
#pragma unroll
      for (int r = 0; r < 8; ++r) {
        p0[r] = st0[r] * kscale;
        p1[r] = st1[r] * kscale;
        bmax = fmaxf(bmax, fmaxf(p0[r], p1[r]));
      }
      bmax = fmaxf(bmax, __shfl_xor(bmax, 16, 32));
      const float m_new = fmaxf(m_run[qt], bmax);
      const float corr = EXP2F(m_run[qt] - m_new);
      float bsum = 0.0f;
#pragma unroll
      for (int r = 0; r < 8; ++r) {
        p0[r] = EXP2F(p0[r] - m_new);
        p1[r] = EXP2F(p1[r] - m_new);
        bsum += p0[r] + p1[r];
      }
      bsum += __shfl_xor(bsum, 16, 32);
      l_run[qt] = l_run[qt] * corr + bsum;
      m_run[qt] = m_new;
#pragma unroll
      for (int t = 0; t < 4; ++t)
#pragma unroll
        for (int e = 0; e < 8; ++e) o[qt][t][e] *= corr;

      // Rebuild P^T into B layout: element e -> key = 16*hi + e.
      // C/D lane-half M-split matches B lane-half K-split -> one xor16 each.
#pragma unroll
      for (int r = 0; r < 8; ++r) {
        const float x0 = __shfl_xor(p0[r], 16, 32);  // partner half, tile0
        const float x1 = __shfl_xor(p1[r], 16, 32);  // partner half, tile1
        pb[qt][r]     = (_Float16)(hi ? x1 : p0[r]);
        pb[qt][8 + r] = (_Float16)(hi ? p1[r] : x0);
      }
    }

    // O^T += V^T * P^T over 4 d-tiles; V A-fragments shared by both q-tiles.
#pragma unroll
    for (int t = 0; t < 4; ++t) {
      v16h va = loadA_f16(Vbh + (size_t)(t * 16) * SEQ + kb, SEQ, lane);
      o[0][t] = wmma_f16(va, pb[0], o[0][t]);
      o[1][t] = wmma_f16(va, pb[1], o[1][t]);
    }
  }

  // Normalize and store to (b, seq, HID) f16, 16B vector stores.
#pragma unroll
  for (int qt = 0; qt < 2; ++qt) {
    const float inv_l = 1.0f / l_run[qt];
    const int q = q0 + qt * 16 + nl;
    const size_t base = ((size_t)(bb * SEQ + q)) * HID + h * DHEAD + hi * 8;
#pragma unroll
    for (int t = 0; t < 4; ++t) {
      v8h ov;
#pragma unroll
      for (int r = 0; r < 8; ++r) ov[r] = (_Float16)(o[qt][t][r] * inv_l);
      *(v8h*)(AOut + base + t * 16) = ov;
    }
  }
}

// ---- Kernel 3: output projection  out = A(f16, 4096x512) * WoT(f16) + bo ----
__global__ void __launch_bounds__(256)
final_gemm(const _Float16* __restrict__ A, const _Float16* __restrict__ WoT,
           const float* __restrict__ bo, float* __restrict__ Out) {
  const int lane = threadIdx.x & 31;
  const int wave = threadIdx.x >> 5;
  const int row0 = blockIdx.x * 256 + wave * 32;
  const int col0 = blockIdx.y * 64;
  const int nl = lane & 15, hi = (lane >> 4) & 1;

  v8f acc[2][4];
#pragma unroll
  for (int rt = 0; rt < 2; ++rt)
#pragma unroll
    for (int t = 0; t < 4; ++t)
#pragma unroll
      for (int e = 0; e < 8; ++e) acc[rt][t][e] = 0.0f;

  for (int kb = 0; kb < HID; kb += 32) {
    v16h a0 = loadA_f16(A + (size_t)row0 * HID + kb, HID, lane);
    v16h a1 = loadA_f16(A + (size_t)(row0 + 16) * HID + kb, HID, lane);
#pragma unroll
    for (int t = 0; t < 4; ++t) {
      v16h b = loadB_kcontig(WoT + (size_t)(col0 + t * 16) * HID + kb, HID, lane);
      acc[0][t] = wmma_f16(a0, b, acc[0][t]);
      acc[1][t] = wmma_f16(a1, b, acc[1][t]);
    }
  }

#pragma unroll
  for (int rt = 0; rt < 2; ++rt) {
#pragma unroll
    for (int t = 0; t < 4; ++t) {
      const int col = col0 + t * 16 + nl;
      const float bias = bo[col];
#pragma unroll
      for (int r = 0; r < 8; ++r) {
        const int g = row0 + rt * 16 + 8 * hi + r;
        Out[(size_t)g * DIM + col] = acc[rt][t][r] + bias;
      }
    }
  }
}

extern "C" void kernel_launch(void* const* d_in, const int* in_sizes, int n_in,
                              void* d_out, int out_size, void* d_ws, size_t ws_size,
                              hipStream_t stream) {
  (void)in_sizes; (void)n_in; (void)out_size; (void)ws_size;
  const float* x   = (const float*)d_in[0];
  const float* ctx = (const float*)d_in[1];
  const float* Wq  = (const float*)d_in[2];
  const float* Wk  = (const float*)d_in[3];
  const float* Wv  = (const float*)d_in[4];
  const float* Wo  = (const float*)d_in[5];
  const float* bo  = (const float*)d_in[6];
  float* out = (float*)d_out;

  char* ws = (char*)d_ws;                    // 20 MB used
  _Float16* qf  = (_Float16*)(ws);                         // (bh, seq, 64)  4 MB
  _Float16* kf  = (_Float16*)(ws + (((size_t)4)  << 20));  // (bh, seq, 64)  4 MB
  _Float16* vt  = (_Float16*)(ws + (((size_t)8)  << 20));  // (bh, 64, seq)  4 MB
  _Float16* ao  = (_Float16*)(ws + (((size_t)12) << 20));  // (b*seq, 512)   4 MB
  _Float16* wqT = (_Float16*)(ws + (((size_t)16) << 20));  // (512, 1024)    1 MB
  _Float16* wkT = (_Float16*)(ws + (((size_t)17) << 20));
  _Float16* wvT = (_Float16*)(ws + (((size_t)18) << 20));
  _Float16* woT = (_Float16*)(ws + (((size_t)19) << 20));  // (1024, 512)    1 MB

  const int tElems = DIM * HID;
  transpose_f32_to_f16<<<(tElems + 255) / 256, 256, 0, stream>>>(Wq, wqT, DIM, HID);
  transpose_f32_to_f16<<<(tElems + 255) / 256, 256, 0, stream>>>(Wk, wkT, DIM, HID);
  transpose_f32_to_f16<<<(tElems + 255) / 256, 256, 0, stream>>>(Wv, wvT, DIM, HID);
  transpose_f32_to_f16<<<(tElems + 255) / 256, 256, 0, stream>>>(Wo, woT, HID, DIM);

  dim3 pgrid(4096 / 256, HID / 64);   // (16, 8)
  proj_gemm<0><<<pgrid, 256, 0, stream>>>(x,   wqT, qf, DIM);
  proj_gemm<0><<<pgrid, 256, 0, stream>>>(ctx, wkT, kf, DIM);
  proj_gemm<1><<<pgrid, 256, 0, stream>>>(ctx, wvT, vt, DIM);

  dim3 agrid(SEQ / 128, BATCH * HEADS);  // (16, 16), 4 waves/block
  attention_kernel<<<agrid, 128, 0, stream>>>(qf, kf, vt, ao);

  dim3 fgrid(4096 / 256, DIM / 64);   // (16, 16)
  final_gemm<<<fgrid, 256, 0, stream>>>(ao, woT, bo, out);
}
